// BipropLinear_30468497998575
// MI455X (gfx1250) — matrix operand order repaired and checked by reading, not compiled, not run on hardware
//
#include <hip/hip_runtime.h>
#include <hip/hip_bf16.h>
#include <stdint.h>

// ---------------- problem constants (from reference) ----------------
#define D_IN   4096
#define D_OUT  4096
#define BATCH  4
#define SEQQ   2048
#define MM     (BATCH * SEQQ)        // 8192 GEMM rows
#define NN     D_OUT                 // 4096 GEMM cols
#define KK     D_IN                  // 4096 GEMM depth
#define KEEPC  3277                  // ceil(0.8 * 4096)
#define NDROP  (D_IN - KEEPC)        // 819

// ---------------- WMMA vector types ----------------
typedef __attribute__((ext_vector_type(16))) __bf16 v16bf;
typedef __attribute__((ext_vector_type(8)))  __bf16 v8bf;
typedef __attribute__((ext_vector_type(8)))  float  v8f;
typedef __attribute__((ext_vector_type(4)))  int    v4i;

static __device__ __forceinline__ unsigned short f32_to_bf16_bits(float f) {
    union { float f; uint32_t u; } v; v.f = f;
    // round-to-nearest-even
    uint32_t r = v.u + 0x7FFFu + ((v.u >> 16) & 1u);
    return (unsigned short)(r >> 16);
}

// ---------------- CDNA5 async global->LDS path (guarded) ----------------
#if defined(__gfx1250__) && __has_builtin(__builtin_amdgcn_global_load_async_to_lds_b128)
#define USE_ASYNC_LDS 1
#define GLOBAL_AS __attribute__((address_space(1)))
#define SHARED_AS __attribute__((address_space(3)))
static __device__ __forceinline__ void async_copy16(const void* g, void* l) {
    __builtin_amdgcn_global_load_async_to_lds_b128(
        (GLOBAL_AS v4i*)g, (SHARED_AS v4i*)l, 0, 0);
}
static __device__ __forceinline__ void async_join() {
#if __has_builtin(__builtin_amdgcn_s_wait_asynccnt)
    __builtin_amdgcn_s_wait_asynccnt(0);
#else
    asm volatile("s_wait_asynccnt 0" ::: "memory");
#endif
}
#else
#define USE_ASYNC_LDS 0
#endif

// ================================================================
// 1) column means of score: cm[c] = mean over rows of score[r][c]
// ================================================================
__global__ void colmean_kernel(const float* __restrict__ score, float* __restrict__ cm) {
    int c = blockIdx.x * 256 + threadIdx.x;          // 16 blocks x 256 = 4096 cols
    float acc = 0.0f;
    for (int r = 0; r < D_OUT; ++r)
        acc += score[(size_t)r * D_IN + c];          // coalesced across threads
    cm[c] = acc * (1.0f / (float)D_OUT);
}

// ================================================================
// 2) rank-based selection == stable argsort drop of NDROP smallest.
// ================================================================
__global__ void mask_select_kernel(const float* __restrict__ cm,
                                   float* __restrict__ mask_col,
                                   float* __restrict__ mask_out,
                                   float* __restrict__ Ssum) {
    __shared__ float sm[D_IN];
    for (int i = threadIdx.x; i < D_IN; i += 1024) sm[i] = cm[i];
    __syncthreads();
    for (int c = threadIdx.x; c < D_IN; c += 1024) {
        float v = sm[c];
        int rank = 0;
        for (int j = 0; j < D_IN; ++j) {
            float u = sm[j];
            rank += (u < v) | ((u == v) & (j < c));   // stable tie-break by index
        }
        float mv = (rank >= NDROP) ? 1.0f : 0.0f;     // dropped iff among NDROP smallest
        mask_col[c] = mv;
        mask_out[c] = mv;                             // reference output #3
    }
    if (threadIdx.x == 0) *Ssum = 0.0f;
}

// ================================================================
// 3) associate_memory = sign(w)*mask (output #2), pack ternary
//    weight TRANSPOSED as bf16 [K][N], accumulate sum(|w|*mask).
// ================================================================
__global__ void binarize_kernel(const float* __restrict__ weight,
                                const float* __restrict__ mask_col,
                                float* __restrict__ assoc,
                                unsigned short* __restrict__ wbT,
                                float* __restrict__ Ssum) {
    int i = blockIdx.x * 256 + threadIdx.x;           // over D_OUT*D_IN
    int o = i >> 12;                                  // row (d_out)
    int c = i & (D_IN - 1);                           // col (d_in)
    float w = weight[i];
    float m = mask_col[c];
    float sg = (w > 0.0f) ? 1.0f : ((w < 0.0f) ? -1.0f : 0.0f);
    float t = sg * m;                                 // in {-1, 0, +1}, exact in bf16
    assoc[i] = t;                                     // reference output #2
    wbT[(size_t)c * NN + o] = f32_to_bf16_bits(t);    // transposed ternary weight
    float contrib = fabsf(w) * m;
    __shared__ float red[256];
    red[threadIdx.x] = contrib;
    __syncthreads();
    #pragma unroll
    for (int s = 128; s > 0; s >>= 1) {
        if (threadIdx.x < s) red[threadIdx.x] += red[threadIdx.x + s];
        __syncthreads();
    }
    if (threadIdx.x == 0) atomicAdd(Ssum, red[0]);
}

// ================================================================
// 4) sample f32 -> bf16 (vectorized 4-wide)
// ================================================================
typedef __attribute__((ext_vector_type(4))) unsigned short v4s;
__global__ void cvt_kernel(const float* __restrict__ s, unsigned short* __restrict__ sb) {
    size_t i = (size_t)blockIdx.x * 256 + threadIdx.x;   // per 4 elements
    float4 v = ((const float4*)s)[i];
    v4s o;
    o.x = f32_to_bf16_bits(v.x);
    o.y = f32_to_bf16_bits(v.y);
    o.z = f32_to_bf16_bits(v.z);
    o.w = f32_to_bf16_bits(v.w);
    ((v4s*)sb)[i] = o;
}

// ================================================================
// 5) GEMM: out = scaling * A(bf16, MxK) * Bt(bf16 ternary, KxN)
//    128x128 block tile, BK=32, 8 waves 4(M)x2(N), 2x4 WMMA tiles
//    per wave. Double-buffered LDS; async global->LDS when available.
// ================================================================
#define BMT 128
#define BNT 128
#define BKT 32
#define LDA_P (BKT + 8)    // 40 shorts -> 80 B row stride (16B aligned)
#define LDB_P (BNT + 8)    // 136 shorts -> 272 B row stride (16B aligned)

__global__ __launch_bounds__(256, 2)
void gemm_wmma_kernel(const unsigned short* __restrict__ A,
                      const unsigned short* __restrict__ Bt,
                      const float* __restrict__ Ssum,
                      float* __restrict__ out) {
    __shared__ __align__(16) unsigned short lA[2][BMT][LDA_P];
    __shared__ __align__(16) unsigned short lB[2][BKT][LDB_P];

    const int tid  = threadIdx.x;
    const int lane = tid & 31;
    const int wave = tid >> 5;
    const int waveM = wave & 3;      // 0..3 -> 32-row slab
    const int waveN = wave >> 2;     // 0..1 -> 64-col slab

    const int m0 = blockIdx.y * BMT;
    const int n0 = blockIdx.x * BNT;

    const float scaling = Ssum[0] * (1.0f / ((float)D_OUT * (float)KEEPC));

    v8f acc[2][4] = {};

    const unsigned short* gA = A  + (size_t)m0 * KK;      // + k0 per stage
    const unsigned short* gB = Bt + n0;                   // + k0*NN per stage

    // per-thread staging coordinates (16 B of A and 16 B of B per pass, 2 passes)
    int ar[2], ac[2], br[2], bc[2];
    #pragma unroll
    for (int p = 0; p < 2; ++p) {
        int lin = (p * 256 + tid) * 8;
        ar[p] = lin >> 5; ac[p] = lin & 31;               // A: 32 shorts per row
        br[p] = lin >> 7; bc[p] = lin & 127;              // B: 128 shorts per row
    }

    auto stage = [&](int kt, int buf) {
        int k0 = kt * BKT;
        #pragma unroll
        for (int p = 0; p < 2; ++p) {
#if USE_ASYNC_LDS
            async_copy16(gA + (size_t)ar[p] * KK + k0 + ac[p], &lA[buf][ar[p]][ac[p]]);
            async_copy16(gB + (size_t)(k0 + br[p]) * NN + bc[p], &lB[buf][br[p]][bc[p]]);
#else
            *(uint4*)&lA[buf][ar[p]][ac[p]] =
                *(const uint4*)(gA + (size_t)ar[p] * KK + k0 + ac[p]);
            *(uint4*)&lB[buf][br[p]][bc[p]] =
                *(const uint4*)(gB + (size_t)(k0 + br[p]) * NN + bc[p]);
#endif
        }
    };

    auto compute = [&](int buf) {
        // A fragments: lane<16 holds M=lane, K {0..7,16..23}; lane>=16 -> K {8..15,24..31}
        v16bf afrag[2];
        #pragma unroll
        for (int mi = 0; mi < 2; ++mi) {
            int row = waveM * 32 + mi * 16 + (lane & 15);
            int kh  = (lane >> 4) * 8;
            union { v16bf v; v8bf h[2]; } u;
            u.h[0] = *(const v8bf*)&lA[buf][row][kh];
            u.h[1] = *(const v8bf*)&lA[buf][row][kh + 16];
            afrag[mi] = u.v;
        }
        // B fragments: lane<16 holds N=lane, K {0..7,16..23}; lane>=16 -> K {8..15,24..31}
        v16bf bfrag[4];
        #pragma unroll
        for (int nj = 0; nj < 4; ++nj) {
            int nb = waveN * 64 + nj * 16 + (lane & 15);
            int kb = (lane >> 4) * 8;
            union { v16bf v; unsigned short e[16]; } u;
            #pragma unroll
            for (int s = 0; s < 16; ++s)
                u.e[s] = lB[buf][kb + (s & 7) + ((s >> 3) * 16)][nb];
            bfrag[nj] = u.v;
        }
        #pragma unroll
        for (int mi = 0; mi < 2; ++mi)
            #pragma unroll
            for (int nj = 0; nj < 4; ++nj)
                acc[mi][nj] = __builtin_amdgcn_wmma_f32_16x16x32_bf16(
                    false, afrag[mi], false, bfrag[nj],
                    (short)0, acc[mi][nj], false, false);
    };

    const int nk = KK / BKT;          // 128 K-steps

    stage(0, 0);
#if USE_ASYNC_LDS
    async_join();
#endif
    __syncthreads();

    for (int kt = 0; kt < nk; ++kt) {
        int cur = kt & 1;
        if (kt + 1 < nk)
            stage(kt + 1, cur ^ 1);   // overlap next-tile copy with compute
        compute(cur);
#if USE_ASYNC_LDS
        async_join();                 // next buffer's async writes landed
#endif
        __syncthreads();              // + all waves done reading `cur`
    }

    // epilogue: D layout is VGPR r -> M = r + 8*(lane>=16), N = lane%16
    #pragma unroll
    for (int mi = 0; mi < 2; ++mi) {
        #pragma unroll
        for (int nj = 0; nj < 4; ++nj) {
            int mt = m0 + waveM * 32 + mi * 16 + ((lane >> 4) * 8);
            int nt = n0 + waveN * 64 + nj * 16 + (lane & 15);
            float* po = out + (size_t)mt * NN + nt;
            #pragma unroll
            for (int r = 0; r < 8; ++r)
                po[(size_t)r * NN] = acc[mi][nj][r] * scaling;
        }
    }
}

// ================================================================
// launch
// ================================================================
extern "C" void kernel_launch(void* const* d_in, const int* in_sizes, int n_in,
                              void* d_out, int out_size, void* d_ws, size_t ws_size,
                              hipStream_t stream) {
    const float* sample = (const float*)d_in[0];   // [4,2048,4096]
    const float* weight = (const float*)d_in[1];   // [4096,4096]
    const float* score  = (const float*)d_in[2];   // [4096,4096]

    float* out      = (float*)d_out;                       // [M,N]
    float* assoc    = out + (size_t)MM * NN;               // [D_OUT,D_IN]
    float* mask_out = assoc + (size_t)D_OUT * D_IN;        // [D_IN]

    char* ws = (char*)d_ws;
    float*          cm       = (float*)(ws);                       // 16 KB
    float*          mask_col = (float*)(ws + (16 << 10));          // 16 KB
    float*          Ssum     = (float*)(ws + (32 << 10));          // 4 B
    unsigned short* sampleB  = (unsigned short*)(ws + (64 << 10)); // 64 MB
    unsigned short* wbT      = sampleB + (size_t)MM * KK;          // 32 MB

    colmean_kernel<<<D_IN / 256, 256, 0, stream>>>(score, cm);
    mask_select_kernel<<<1, 1024, 0, stream>>>(cm, mask_col, mask_out, Ssum);
    binarize_kernel<<<(D_OUT * D_IN) / 256, 256, 0, stream>>>(weight, mask_col, assoc, wbT, Ssum);
    cvt_kernel<<<((size_t)MM * KK / 4) / 256, 256, 0, stream>>>(sample, sampleB);

    dim3 grid(NN / BNT, MM / BMT);   // (32, 64)
    gemm_wmma_kernel<<<grid, 256, 0, stream>>>(sampleB, wbT, Ssum, out);
}
